// MultiHeadSelfAttention_76355928588897
// MI455X (gfx1250) — matrix operand order repaired and checked by reading, not compiled, not run on hardware
//
#include <hip/hip_runtime.h>
#include <hip/hip_bf16.h>

// ---------------------------------------------------------------------------
// MHSA forward for MI455X (gfx1250, wave32, WMMA bf16 16x16x32, f32 accum)
// B=2, S=2048, D=2048, H=16, HD=128
// ---------------------------------------------------------------------------

#define B_ 2
#define S_ 2048
#define D_ 2048
#define H_ 16
#define HD_ 128
#define NX_ (B_ * S_ * D_)   // 8388608 elements
#define NW_ (D_ * D_)        // 4194304 elements

typedef __attribute__((ext_vector_type(16))) __bf16 v16bf;
typedef __attribute__((ext_vector_type(8)))  __bf16 bf16x8;
typedef __attribute__((ext_vector_type(8)))  float  v8f;

union FragU { v16bf v; bf16x8 h[2]; };

// A-fragment (16x32, M x K), 16-bit, row-major source with leading dim ld.
// Lane L (lo=L&15, hi=L>>4): row = row0+lo, K chunks [hi*8, hi*8+8) and
// [16+hi*8, 16+hi*8+8)  (per ISA 16-bit A-matrix table).
__device__ __forceinline__ v16bf load_frag_a(const __bf16* __restrict__ p,
                                             int ld, int row0, int k0, int lane) {
  int lo = lane & 15, hi = lane >> 4;
  const __bf16* base = p + (size_t)(row0 + lo) * ld + k0;
  FragU f;
  f.h[0] = *(const bf16x8*)(base + hi * 8);
  f.h[1] = *(const bf16x8*)(base + 16 + hi * 8);
  return f.v;
}

// B-fragment (32x16, K x N) where B[k][n] comes from a row-major matrix whose
// row index is n (i.e. we multiply by its transpose): lane lo holds column
// n=row0+lo, K chunks [hi*8, hi*8+8) and [16+hi*8, 16+hi*8+8) (matches A).
__device__ __forceinline__ v16bf load_frag_b(const __bf16* __restrict__ p,
                                             int ld, int row0, int k0, int lane) {
  int lo = lane & 15, hi = lane >> 4;
  const __bf16* base = p + (size_t)(row0 + lo) * ld + k0 + hi * 8;
  FragU f;
  f.h[0] = *(const bf16x8*)(base);
  f.h[1] = *(const bf16x8*)(base + 16);
  return f.v;
}

// ---------------------------------------------------------------------------
// f32 -> bf16 conversion (vectorized x4)
// ---------------------------------------------------------------------------
__global__ void cvt_f32_bf16(const float* __restrict__ in,
                             __bf16* __restrict__ out, int n) {
  int i = (blockIdx.x * blockDim.x + threadIdx.x) * 4;
  if (i + 3 < n) {
    float4 v = *(const float4*)(in + i);
    out[i + 0] = (__bf16)v.x;
    out[i + 1] = (__bf16)v.y;
    out[i + 2] = (__bf16)v.z;
    out[i + 3] = (__bf16)v.w;
  }
}

// ---------------------------------------------------------------------------
// GEMM: C[m,n] = sum_k A[m,k] * W[n,k] + bias[n]       (x @ W^T + b)
// Each wave computes a 32x64 tile. Per K-step of 32: all 6 fragments (2 A,
// 4 B) are fetched up-front as one clause, then 8 WMMAs drain with staggered
// partial waits. grid.x = M/32, grid.y = N/256, block = 128 (4 waves).
// mode 0: Q  -> bf16 [B,H,S,HD], pre-scaled by 1/sqrt(HD)
// mode 1: K  -> bf16 [B,H,S,HD] + f32 [B,H,S,HD] (d_out)
// mode 2: V  -> bf16 [B,H,HD,S] (transposed) + f32 [B,H,S,HD] (d_out)
// mode 3: O  -> f32 [B,S,D] (d_out)
// ---------------------------------------------------------------------------
__global__ void __launch_bounds__(128)
gemm_wmma(const __bf16* __restrict__ A, const __bf16* __restrict__ W,
          const float* __restrict__ bias, __bf16* __restrict__ out_bf,
          float* __restrict__ out_f32, int mode) {
  const int lane = threadIdx.x & 31;
  const int wave = threadIdx.x >> 5;
  const int m0 = blockIdx.x * 32;
  const int n0 = blockIdx.y * 256 + wave * 64;
  const int lo = lane & 15, hi = lane >> 4;

  v8f acc[2][4] = {};
  for (int k0 = 0; k0 < D_; k0 += 32) {
    // Unconditional speculative prefetch of the weight stream 4 K-blocks
    // ahead (global_prefetch_b8; OOB prefetches are silently dropped).
    __builtin_prefetch(W + (size_t)(n0 + lo) * D_ + k0 + 128, 0, 1);
    __builtin_prefetch(W + (size_t)(n0 + 16 + lo) * D_ + k0 + 128, 0, 1);

    // fetch all fragments for this K-step up front (single load clause)
    v16bf a0 = load_frag_a(A, D_, m0, k0, lane);
    v16bf a1 = load_frag_a(A, D_, m0 + 16, k0, lane);
    v16bf b0 = load_frag_b(W, D_, n0, k0, lane);
    v16bf b1 = load_frag_b(W, D_, n0 + 16, k0, lane);
    v16bf b2 = load_frag_b(W, D_, n0 + 32, k0, lane);
    v16bf b3 = load_frag_b(W, D_, n0 + 48, k0, lane);

    acc[0][0] = __builtin_amdgcn_wmma_f32_16x16x32_bf16(
        false, a0, false, b0, (short)0, acc[0][0], false, false);
    acc[1][0] = __builtin_amdgcn_wmma_f32_16x16x32_bf16(
        false, a1, false, b0, (short)0, acc[1][0], false, false);
    acc[0][1] = __builtin_amdgcn_wmma_f32_16x16x32_bf16(
        false, a0, false, b1, (short)0, acc[0][1], false, false);
    acc[1][1] = __builtin_amdgcn_wmma_f32_16x16x32_bf16(
        false, a1, false, b1, (short)0, acc[1][1], false, false);
    acc[0][2] = __builtin_amdgcn_wmma_f32_16x16x32_bf16(
        false, a0, false, b2, (short)0, acc[0][2], false, false);
    acc[1][2] = __builtin_amdgcn_wmma_f32_16x16x32_bf16(
        false, a1, false, b2, (short)0, acc[1][2], false, false);
    acc[0][3] = __builtin_amdgcn_wmma_f32_16x16x32_bf16(
        false, a0, false, b3, (short)0, acc[0][3], false, false);
    acc[1][3] = __builtin_amdgcn_wmma_f32_16x16x32_bf16(
        false, a1, false, b3, (short)0, acc[1][3], false, false);
  }

  const float qscale = 0.08838834764831845f;  // 1/sqrt(128)
#pragma unroll
  for (int u = 0; u < 2; ++u) {
#pragma unroll
    for (int t = 0; t < 4; ++t) {
#pragma unroll
      for (int r = 0; r < 8; ++r) {
        int m = m0 + u * 16 + r + 8 * hi;
        int n = n0 + t * 16 + lo;
        float val = acc[u][t][r] + bias[n];
        int b_ = m >> 11, s_ = m & (S_ - 1);
        int h_ = n >> 7, hd = n & (HD_ - 1);
        size_t bhsd = (((size_t)b_ * H_ + h_) * S_ + s_) * HD_ + hd;
        if (mode == 0) {
          out_bf[bhsd] = (__bf16)(val * qscale);
        } else if (mode == 1) {
          out_bf[bhsd] = (__bf16)val;
          out_f32[bhsd] = val;
        } else if (mode == 2) {
          out_bf[(((size_t)b_ * H_ + h_) * HD_ + hd) * S_ + s_] = (__bf16)val;
          out_f32[bhsd] = val;
        } else {
          out_f32[(size_t)m * D_ + n] = val;
        }
      }
    }
  }
}

// ---------------------------------------------------------------------------
// Causal flash attention, one wave per 16-row query block.
// Q already scaled by 1/sqrt(HD). KV processed in 32-wide blocks.
// qb: [B,H,S,HD] bf16   kb: [B,H,S,HD] bf16   vtb: [B,H,HD,S] bf16
// ab: [B,S,D] bf16 (attention output, consumed by O-projection)
// grid.x = S/16, grid.y = B*H, block = 32
// ---------------------------------------------------------------------------
__global__ void __launch_bounds__(32)
attn_wmma(const __bf16* __restrict__ qb, const __bf16* __restrict__ kb,
          const __bf16* __restrict__ vtb, __bf16* __restrict__ ab) {
  const int lane = threadIdx.x;
  const int lo = lane & 15, hi = lane >> 4;
  const int bh = blockIdx.y;             // b*H + h
  const int q0 = blockIdx.x * 16;

  const __bf16* Q  = qb  + (size_t)bh * S_ * HD_;
  const __bf16* K  = kb  + (size_t)bh * S_ * HD_;
  const __bf16* VT = vtb + (size_t)bh * HD_ * S_;

  __shared__ float  s_sc[16 * 32];
  __shared__ __bf16 s_p[16 * 32];
  __shared__ float  s_bcast[16];

  // Q fragments for the whole head dim (4 chunks of K=32)
  v16bf qf[4];
#pragma unroll
  for (int c = 0; c < 4; ++c) qf[c] = load_frag_a(Q, HD_, q0, c * 32, lane);

  v8f acc[8] = {};
  float m_i = -3.0e38f;   // running row max (valid in lanes 0..15, row = lane)
  float l_i = 0.0f;       // running row sum

  for (int kv0 = 0; kv0 < q0 + 16; kv0 += 32) {
    // ---- scores: S = Q K^T, two 16x16 tiles covering kv cols [kv0, kv0+32)
    v8f sc0 = {}, sc1 = {};
#pragma unroll
    for (int c = 0; c < 4; ++c) {
      v16bf kf0 = load_frag_b(K, HD_, kv0, c * 32, lane);
      sc0 = __builtin_amdgcn_wmma_f32_16x16x32_bf16(false, qf[c], false, kf0,
                                                    (short)0, sc0, false, false);
      v16bf kf1 = load_frag_b(K, HD_, kv0 + 16, c * 32, lane);
      sc1 = __builtin_amdgcn_wmma_f32_16x16x32_bf16(false, qf[c], false, kf1,
                                                    (short)0, sc1, false, false);
    }
    // spill scores to LDS (C layout: N = lo, M = r + 8*hi)
#pragma unroll
    for (int r = 0; r < 8; ++r) {
      s_sc[(r + 8 * hi) * 32 + lo]      = sc0[r];
      s_sc[(r + 8 * hi) * 32 + 16 + lo] = sc1[r];
    }
    __syncthreads();

    // ---- online softmax: lanes 0..15 each own one query row
    if (lane < 16) {
      const int row = lane, srow = q0 + row;
      float mnew = m_i;
      for (int c = 0; c < 32; ++c) {
        float xv = (kv0 + c <= srow) ? s_sc[row * 32 + c] : -3.0e38f;
        mnew = fmaxf(mnew, xv);
      }
      float corr = __expf(m_i - mnew);
      float sum = 0.0f;
      for (int c = 0; c < 32; ++c) {
        float xv = (kv0 + c <= srow) ? s_sc[row * 32 + c] : -3.0e38f;
        float p = __expf(xv - mnew);
        sum += p;
        s_p[row * 32 + c] = (__bf16)p;
      }
      l_i = l_i * corr + sum;
      m_i = mnew;
      s_bcast[row] = corr;
    }
    __syncthreads();

    // ---- rescale accumulators by per-row correction factor
    float cr[8];
#pragma unroll
    for (int r = 0; r < 8; ++r) cr[r] = s_bcast[r + 8 * hi];
#pragma unroll
    for (int t = 0; t < 8; ++t)
#pragma unroll
      for (int r = 0; r < 8; ++r) acc[t][r] *= cr[r];

    // ---- PV: P (16x32) from LDS as A-fragment, V^T rows as B-fragments
    v16bf pf = load_frag_a(s_p, 32, 0, 0, lane);
#pragma unroll
    for (int t = 0; t < 8; ++t) {
      v16bf vf = load_frag_b(VT, S_, t * 16, kv0, lane);
      acc[t] = __builtin_amdgcn_wmma_f32_16x16x32_bf16(
          false, pf, false, vf, (short)0, acc[t], false, false);
    }
    __syncthreads();
  }

  // ---- finalize: divide by l, write to attention output [B,S,D] as bf16
  if (lane < 16) s_bcast[lane] = 1.0f / l_i;
  __syncthreads();
  float inv[8];
#pragma unroll
  for (int r = 0; r < 8; ++r) inv[r] = s_bcast[r + 8 * hi];

  const int b_ = bh >> 4, h_ = bh & 15;
#pragma unroll
  for (int t = 0; t < 8; ++t) {
#pragma unroll
    for (int r = 0; r < 8; ++r) {
      int srow = q0 + r + 8 * hi;
      int d = h_ * HD_ + t * 16 + lo;
      ab[((size_t)b_ * S_ + srow) * D_ + d] = (__bf16)(acc[t][r] * inv[r]);
    }
  }
}

// ---------------------------------------------------------------------------
// Host-side launch
// ---------------------------------------------------------------------------
extern "C" void kernel_launch(void* const* d_in, const int* in_sizes, int n_in,
                              void* d_out, int out_size, void* d_ws, size_t ws_size,
                              hipStream_t stream) {
  const float* x  = (const float*)d_in[0];
  const float* Wq = (const float*)d_in[1];
  const float* bq = (const float*)d_in[2];
  const float* Wk = (const float*)d_in[3];
  const float* bk = (const float*)d_in[4];
  const float* Wv = (const float*)d_in[5];
  const float* bv = (const float*)d_in[6];
  const float* Wo = (const float*)d_in[7];
  const float* bo = (const float*)d_in[8];

  float* out  = (float*)d_out;            // [B,S,D]
  float* kout = out + NX_;                // [B,H,S,HD]
  float* vout = out + 2 * (size_t)NX_;    // [B,H,S,HD]

  // workspace carve-up (bf16)
  __bf16* xb  = (__bf16*)d_ws;
  __bf16* wqb = xb  + NX_;
  __bf16* wkb = wqb + NW_;
  __bf16* wvb = wkb + NW_;
  __bf16* wob = wvb + NW_;
  __bf16* qb  = wob + NW_;   // [B,H,S,HD], pre-scaled
  __bf16* kb  = qb  + NX_;   // [B,H,S,HD]
  __bf16* vtb = kb  + NX_;   // [B,H,HD,S]
  __bf16* ab  = vtb + NX_;   // [B,S,D] attention output

  // 1) convert inputs to bf16
  cvt_f32_bf16<<<NX_ / 1024, 256, 0, stream>>>(x, xb, NX_);
  cvt_f32_bf16<<<NW_ / 1024, 256, 0, stream>>>(Wq, wqb, NW_);
  cvt_f32_bf16<<<NW_ / 1024, 256, 0, stream>>>(Wk, wkb, NW_);
  cvt_f32_bf16<<<NW_ / 1024, 256, 0, stream>>>(Wv, wvb, NW_);
  cvt_f32_bf16<<<NW_ / 1024, 256, 0, stream>>>(Wo, wob, NW_);

  // 2) QKV projections (M = B*S = 4096, N = D = 2048)
  dim3 ggrid(B_ * S_ / 32, D_ / 256);
  gemm_wmma<<<ggrid, 128, 0, stream>>>(xb, wqb, bq, qb, nullptr, 0);
  gemm_wmma<<<ggrid, 128, 0, stream>>>(xb, wkb, bk, kb, kout, 1);
  gemm_wmma<<<ggrid, 128, 0, stream>>>(xb, wvb, bv, vtb, vout, 2);

  // 3) causal flash attention
  attn_wmma<<<dim3(S_ / 16, B_ * H_), 32, 0, stream>>>(qb, kb, vtb, ab);

  // 4) output projection -> d_out
  gemm_wmma<<<ggrid, 128, 0, stream>>>(ab, wob, bo, nullptr, out, 3);
}
